// Block_7361573945782
// MI455X (gfx1250) — matrix operand order repaired and checked by reading, not compiled, not run on hardware
//
#include <hip/hip_runtime.h>

// ---------------- problem constants ----------------
constexpr int cB = 4, cH = 32, cW = 32, cDIM = 128;
constexpr int cDI = 256, cN = 16, cR = 8, cK = 4;
constexpr int cL = cH * cW;            // 1024
constexpr int cC = cR + 2 * cN;        // 40 (padded to 48 for WMMA)
constexpr int cCP = 48;
constexpr int cKX = cW / 2 + 1;        // 17 rfft bins
constexpr int cPIX = cH * cKX;         // 544 freq pixels per (b)
constexpr int cM2 = cB * cPIX;         // 2176 freq rows
constexpr float PI2 = 6.28318530717958647692f;

typedef __bf16 bf16_t;
typedef __attribute__((ext_vector_type(16))) bf16_t v16bf;
typedef __attribute__((ext_vector_type(8)))  float  v8f;

static __device__ __forceinline__ float bf2f(bf16_t b) {
    unsigned short s = __builtin_bit_cast(unsigned short, b);
    unsigned u = ((unsigned)s) << 16;
    return __builtin_bit_cast(float, u);
}
static __device__ __forceinline__ bf16_t f2bf(float f) {
    unsigned u = __builtin_bit_cast(unsigned, f);
    u += 0x7FFFu + ((u >> 16) & 1u);
    unsigned short s = (unsigned short)(u >> 16);
    return __builtin_bit_cast(bf16_t, s);
}
static __device__ __forceinline__ float sigm(float x) { return 1.0f / (1.0f + __expf(-x)); }

// ---------------- generic bf16 WMMA GEMM, NT n-tiles per wave ----------------
// C[M,N] = A[M,K] * Bt[N,K]^T  (+bias) (+activation)
// A row-major bf16 (ld=K), Bt row-major bf16 (ld=K) -> natural (O,C) weight layout.
// Each wave computes a 16 x (16*NT) strip: one A fragment feeds NT back-to-back
// WMMAs per K-step (amortizes A loads; 4x wmma per loop vs 1-tile version).
// epi: 0 = none, 1 = +bias, 2 = +bias then leaky_relu(0.1)
template <int NT>
__global__ void k_gemm_wmma(const bf16_t* __restrict__ A, const bf16_t* __restrict__ Bt,
                            const float* __restrict__ bias, float* __restrict__ C,
                            int M, int N, int Kd, int epi) {
    const int lane = threadIdx.x & 31;
    const int wave = threadIdx.x >> 5;
    const int ngroups = N / (16 * NT);
    const int totalTiles = (M >> 4) * ngroups;
    const int tile = blockIdx.x * 8 + wave;
    if (tile >= totalTiles) return;
    const int tm  = (tile / ngroups) << 4;
    const int tng = (tile % ngroups) * (16 * NT);
    const int half = lane >> 4;        // which half-wave
    const int idx16 = lane & 15;
    const int mrow = tm + idx16;       // A row for this lane

    v8f acc[NT];
#pragma unroll
    for (int t = 0; t < NT; ++t) acc[t] = (v8f){};

    for (int k0 = 0; k0 < Kd; k0 += 32) {
        const bf16_t* ap = A + (size_t)mrow * Kd + k0 + half * 16;
        if (k0 + 32 < Kd) __builtin_prefetch(ap + 32, 0, 1);   // global_prefetch_b8
        v16bf a;
#pragma unroll
        for (int j = 0; j < 8; ++j) a[j] = ap[j];              // K = k0+16h .. +7
#pragma unroll
        for (int j = 0; j < 8; ++j) a[8 + j] = ap[8 + j];      // K = k0+16h+8 .. +15

        v16bf b[NT];
#pragma unroll
        for (int t = 0; t < NT; ++t) {
            const bf16_t* bp = Bt + (size_t)(tng + t * 16 + idx16) * Kd + k0 + half * 16;
#pragma unroll
            for (int j = 0; j < 16; ++j) b[t][j] = bp[j];      // K = k0+16h+j
        }
#pragma unroll
        for (int t = 0; t < NT; ++t)
            acc[t] = __builtin_amdgcn_wmma_f32_16x16x32_bf16(false, a, false, b[t],
                                                             (short)0, acc[t], false, false);
    }
    const int mbase = tm + half * 8;
#pragma unroll
    for (int t = 0; t < NT; ++t) {
        const int ncol = tng + t * 16 + idx16;
        const float bv = bias ? bias[ncol] : 0.0f;
#pragma unroll
        for (int r = 0; r < 8; ++r) {
            float v = acc[t][r];
            if (epi >= 1) v += bv;
            if (epi == 2) v = (v > 0.0f) ? v : 0.1f * v;
            C[(size_t)(mbase + r) * N + ncol] = v;
        }
    }
}

// ---------------- weight conversion ----------------
__global__ void k_cvt_bf(const float* __restrict__ s, bf16_t* __restrict__ d, int n) {
    int i = blockIdx.x * blockDim.x + threadIdx.x;
    if (i < n) d[i] = f2bf(s[i]);
}
__global__ void k_cvt_xproj(const float* __restrict__ s, bf16_t* __restrict__ d) {
    int i = blockIdx.x * blockDim.x + threadIdx.x;    // (k, cpad, d)
    if (i >= cK * cCP * cDI) return;
    int dd = i & 255;
    int c = (i >> 8) % cCP;
    int k = i / (cCP * cDI);
    d[i] = (c < cC) ? f2bf(s[((size_t)k * cC + c) * cDI + dd]) : f2bf(0.0f);
}

// ---------------- layernorm variants ----------------
template <int F>
__global__ void k_ln_bf(const float* __restrict__ X, const float* __restrict__ w,
                        const float* __restrict__ bb, bf16_t* __restrict__ out) {
    int row = blockIdx.x, c = threadIdx.x;
    __shared__ float red[F];
    float v = X[(size_t)row * F + c];
    red[c] = v; __syncthreads();
    for (int s = F / 2; s > 0; s >>= 1) { if (c < s) red[c] += red[c + s]; __syncthreads(); }
    float mean = red[0] * (1.0f / F); __syncthreads();
    float dv = v - mean;
    red[c] = dv * dv; __syncthreads();
    for (int s = F / 2; s > 0; s >>= 1) { if (c < s) red[c] += red[c + s]; __syncthreads(); }
    float var = red[0] * (1.0f / F);
    out[(size_t)row * F + c] = f2bf(dv * rsqrtf(var + 1e-5f) * w[c] + bb[c]);
}
// LN of att rows (F=128), transposed write to NCHW f32 (for FFT + final gate)
__global__ void k_ln_t(const float* __restrict__ X, const float* __restrict__ w,
                       const float* __restrict__ bb, float* __restrict__ xnc) {
    int row = blockIdx.x, c = threadIdx.x;            // row = b*1024 + l
    __shared__ float red[cDIM];
    float v = X[(size_t)row * cDIM + c];
    red[c] = v; __syncthreads();
    for (int s = 64; s > 0; s >>= 1) { if (c < s) red[c] += red[c + s]; __syncthreads(); }
    float mean = red[0] * (1.0f / cDIM); __syncthreads();
    float dv = v - mean;
    red[c] = dv * dv; __syncthreads();
    for (int s = 64; s > 0; s >>= 1) { if (c < s) red[c] += red[c + s]; __syncthreads(); }
    float var = red[0] * (1.0f / cDIM);
    float y = dv * rsqrtf(var + 1e-5f) * w[c] + bb[c];
    int b = row >> 10, l = row & 1023;
    xnc[((size_t)b * cDIM + c) * cL + l] = y;
}

// ---------------- depthwise 3x3 conv + SiLU + mask -> NCHW ----------------
__global__ void k_dwconv(const float* __restrict__ xz, const float* __restrict__ cw,
                         const float* __restrict__ cb, const float* __restrict__ mask,
                         float* __restrict__ xc) {
    int idx = blockIdx.x * blockDim.x + threadIdx.x;  // ((b*256+d)*32+h)*32+w
    if (idx >= cB * cDI * cL) return;
    int w = idx & 31, h = (idx >> 5) & 31, d = (idx >> 10) & 255, b = idx >> 18;
    float acc = cb[d];
#pragma unroll
    for (int ky = 0; ky < 3; ++ky) {
        int hh = h + ky - 1;
        if (hh < 0 || hh >= cH) continue;
#pragma unroll
        for (int kx = 0; kx < 3; ++kx) {
            int ww = w + kx - 1;
            if (ww < 0 || ww >= cW) continue;
            acc += xz[((size_t)b * cL + hh * cW + ww) * (2 * cDI) + d] * cw[d * 9 + ky * 3 + kx];
        }
    }
    acc = acc * sigm(acc);                              // SiLU
    acc *= mask[b * cL + h * cW + w];
    xc[idx] = acc;
}

// ---------------- build 4-direction xs (bf16, (k,b,l,d)) ----------------
__global__ void k_build_xs(const float* __restrict__ xc, bf16_t* __restrict__ xs) {
    int idx = blockIdx.x * blockDim.x + threadIdx.x;   // ((k*4+b)*1024+l)*256+d
    if (idx >= cK * cB * cL * cDI) return;
    int d = idx & 255, l = (idx >> 8) & 1023, b = (idx >> 18) & 3, k = idx >> 20;
    int pos;
    if (k == 0)      pos = l;
    else if (k == 1) pos = (l & 31) * cW + (l >> 5);            // transpose scan order
    else if (k == 2) pos = cL - 1 - l;                          // flipped
    else { int l2 = cL - 1 - l; pos = (l2 & 31) * cW + (l2 >> 5); }
    xs[idx] = f2bf(xc[((size_t)b * cDI + d) * cL + pos]);
}

// ---------------- selective scan: block = (b,k), thread = channel d ----------------
__global__ void k_scan(const float* __restrict__ dbl, const bf16_t* __restrict__ xs,
                       const float* __restrict__ dt_w, const float* __restrict__ dt_b,
                       const float* __restrict__ A_log, const float* __restrict__ Ds,
                       float* __restrict__ ys) {
    const int bk = blockIdx.x;                    // k*4 + b
    const int k = bk >> 2;
    const int d = threadIdx.x;                    // 0..255
    float dtw[cR];
#pragma unroll
    for (int r = 0; r < cR; ++r) dtw[r] = dt_w[((size_t)k * cDI + d) * cR + r];
    const float dtb = dt_b[k * cDI + d];
    float Arow[cN];
#pragma unroll
    for (int n = 0; n < cN; ++n) Arow[n] = -__expf(A_log[((size_t)k * cDI + d) * cN + n]);
    const float Dv = Ds[k * cDI + d];
    float h[cN];
#pragma unroll
    for (int n = 0; n < cN; ++n) h[n] = 0.0f;

    const float*  dblk = dbl + (size_t)bk * cL * cCP;
    const bf16_t* xsk  = xs  + (size_t)bk * cL * cDI;
    float*        ysk  = ys  + (size_t)bk * cL * cDI;

    __shared__ float cb[8][cCP];                  // 8-step staging of (dts,B,C)
    for (int l0 = 0; l0 < cL; l0 += 8) {
        for (int i = d; i < 8 * cCP; i += cDI)
            cb[i / cCP][i % cCP] = dblk[(size_t)(l0 + i / cCP) * cCP + (i % cCP)];
        __syncthreads();
#pragma unroll 1
        for (int s = 0; s < 8; ++s) {
            const int l = l0 + s;
            const float u = bf2f(xsk[(size_t)l * cDI + d]);
            float delta = dtb;
#pragma unroll
            for (int r = 0; r < cR; ++r) delta += cb[s][r] * dtw[r];
            delta = (delta > 20.0f) ? delta : log1pf(__expf(delta));   // softplus
            const float du = delta * u;
            float y = 0.0f;
#pragma unroll
            for (int n = 0; n < cN; ++n) {
                h[n] = __expf(delta * Arow[n]) * h[n] + du * cb[s][cR + n];
                y += h[n] * cb[s][cR + cN + n];
            }
            ysk[(size_t)l * cDI + d] = y + Dv * u;
        }
        __syncthreads();
    }
}

// ---------------- combine 4 directions + LN(onorm) * SiLU(z) -> bf16 ----------------
__global__ void k_combine(const float* __restrict__ ys, const float* __restrict__ xz,
                          const float* __restrict__ onw, const float* __restrict__ onb,
                          bf16_t* __restrict__ yact) {
    const int row = blockIdx.x;                   // b*1024 + l
    const int b = row >> 10, l = row & 1023;
    const int d = threadIdx.x;
    const int h = l >> 5, w = l & 31;
    const int lt = w * cH + h;
    const size_t stride = (size_t)cL * cDI;
    float v = ys[((size_t)(0 * cB + b)) * stride + (size_t)l * cDI + d]
            + ys[((size_t)(1 * cB + b)) * stride + (size_t)lt * cDI + d]
            + ys[((size_t)(2 * cB + b)) * stride + (size_t)(cL - 1 - l) * cDI + d]
            + ys[((size_t)(3 * cB + b)) * stride + (size_t)(cL - 1 - lt) * cDI + d];
    __shared__ float red[cDI];
    red[d] = v; __syncthreads();
    for (int s = cDI / 2; s > 0; s >>= 1) { if (d < s) red[d] += red[d + s]; __syncthreads(); }
    float mean = red[0] * (1.0f / cDI); __syncthreads();
    float dv = v - mean;
    red[d] = dv * dv; __syncthreads();
    for (int s = cDI / 2; s > 0; s >>= 1) { if (d < s) red[d] += red[d + s]; __syncthreads(); }
    float var = red[0] * (1.0f / cDI);
    float y = dv * rsqrtf(var + 1e-5f) * onw[d] + onb[d];
    float z = xz[(size_t)row * (2 * cDI) + cDI + d];
    yact[(size_t)row * cDI + d] = f2bf(y * (z * sigm(z)));
}

__global__ void k_add(const float* __restrict__ a, const float* __restrict__ b,
                      float* __restrict__ o, int n) {
    int i = blockIdx.x * blockDim.x + threadIdx.x;
    if (i < n) o[i] = a[i] + b[i];
}

// ---------------- FFT branch ----------------
// rfft along W: xnc (b*128+c, 32, 32) real -> Fr/Fi (img, h, kx)
__global__ void k_rfft_w(const float* __restrict__ xnc, float* __restrict__ Fr,
                         float* __restrict__ Fi) {
    int idx = blockIdx.x * blockDim.x + threadIdx.x;
    if (idx >= cB * cDIM * cH * cKX) return;
    int kx = idx % cKX, r = idx / cKX, h = r & 31, img = r >> 5;
    const float* rp = xnc + ((size_t)img << 10) + (h << 5);
    float step = -(PI2 / cW) * kx, sr = 0.0f, si = 0.0f;
    for (int w = 0; w < cW; ++w) {
        float ang = step * w;
        sr += rp[w] * __cosf(ang);
        si += rp[w] * __sinf(ang);
    }
    Fr[idx] = sr; Fi[idx] = si;
}
// full FFT along H, then mag/phase in (row=(b,ky,kx), col=c) layout + bf16 copies
__global__ void k_fft_h_magpha(const float* __restrict__ Fr, const float* __restrict__ Fi,
                               float* __restrict__ mag, float* __restrict__ pha,
                               bf16_t* __restrict__ magb, bf16_t* __restrict__ phab) {
    int idx = blockIdx.x * blockDim.x + threadIdx.x;  // ((b*128+c)*32+ky)*17+kx
    if (idx >= cB * cDIM * cH * cKX) return;
    int kx = idx % cKX, r = idx / cKX, ky = r & 31, c = (r >> 5) & 127, b = r >> 12;
    float step = -(PI2 / cH) * ky, gr = 0.0f, gi = 0.0f;
    for (int h = 0; h < cH; ++h) {
        size_t s = (((size_t)b * cDIM + c) * cH + h) * cKX + kx;
        float fr = Fr[s], fi = Fi[s];
        float ang = step * h, cs = __cosf(ang), sn = __sinf(ang);
        gr += fr * cs - fi * sn;
        gi += fr * sn + fi * cs;
    }
    float m = sqrtf(gr * gr + gi * gi);
    float p = atan2f(gi, gr);
    size_t o = ((size_t)b * cPIX + ky * cKX + kx) * cDIM + c;
    mag[o] = m; pha[o] = p; magb[o] = f2bf(m); phab[o] = f2bf(p);
}
// stdmean over 64 channels, then x + sigmoid(x)*x -> bf16 GEMM input
__global__ void k_stdmean(const float* __restrict__ t, bf16_t* __restrict__ out) {
    int row = blockIdx.x * blockDim.x + threadIdx.x;
    if (row >= cM2) return;
    const float* x = t + (size_t)row * 64;
    float s = 0.0f, sq = 0.0f;
    for (int c = 0; c < 64; ++c) { s += x[c]; sq += x[c] * x[c]; }
    float mean = s * (1.0f / 64.0f);
    float var = (sq - 64.0f * mean * mean) * (1.0f / 63.0f);
    float sd = sqrtf(fmaxf(var, 0.0f));
    float fs = 0.0f, cnt = 0.0f;
    for (int c = 0; c < 64; ++c) { float f = (x[c] > mean) ? x[c] : 0.0f; fs += f; cnt += (f > 0.0f) ? 1.0f : 0.0f; }
    if (cnt == 0.0f) cnt = 1.0f;
    float above = fs / cnt, inv = 1.0f / (sd + 1e-10f);
    for (int c = 0; c < 64; ++c) {
        float y = (x[c] - above) * inv;
        out[(size_t)row * 64 + c] = f2bf(sigm(y) * y + y);
    }
}
// (mag+dm)*e^{i(pha+dp)}
__global__ void k_recon(const float* __restrict__ mag, const float* __restrict__ magp,
                        const float* __restrict__ pha, const float* __restrict__ phap,
                        float* __restrict__ F2r, float* __restrict__ F2i) {
    int i = blockIdx.x * blockDim.x + threadIdx.x;
    if (i >= cM2 * cDIM) return;
    float m = mag[i] + magp[i], p = pha[i] + phap[i];
    F2r[i] = m * __cosf(p);
    F2i[i] = m * __sinf(p);
}
// inverse FFT along H (complex), /32
__global__ void k_ifft_h(const float* __restrict__ F2r, const float* __restrict__ F2i,
                         float* __restrict__ Hr, float* __restrict__ Hi) {
    int idx = blockIdx.x * blockDim.x + threadIdx.x;  // (b,h,kx,c)
    if (idx >= cB * cH * cKX * cDIM) return;
    int c = idx & 127, r = idx >> 7, kx = r % cKX; r /= cKX;
    int h = r & 31, b = r >> 5;
    float step = (PI2 / cH) * h, hr = 0.0f, hi = 0.0f;
    for (int ky = 0; ky < cH; ++ky) {
        size_t s = ((size_t)b * cPIX + ky * cKX + kx) * cDIM + c;
        float fr = F2r[s], fi = F2i[s];
        float ang = step * ky, cs = __cosf(ang), sn = __sinf(ang);
        hr += fr * cs - fi * sn;
        hi += fr * sn + fi * cs;
    }
    size_t o = ((size_t)b * cPIX + h * cKX + kx) * cDIM + c;
    Hr[o] = hr * (1.0f / 32.0f);
    Hi[o] = hi * (1.0f / 32.0f);
}
// irfft along W with hermitian symmetry, /32 -> sp rows=(b,h,w), cols=c (bf16)
__global__ void k_irfft_w(const float* __restrict__ Hr, const float* __restrict__ Hi,
                          bf16_t* __restrict__ sp) {
    int idx = blockIdx.x * blockDim.x + threadIdx.x;  // (b,h,w,c)
    if (idx >= cB * cH * cW * cDIM) return;
    int c = idx & 127, w = (idx >> 7) & 31, h = (idx >> 12) & 31, b = idx >> 17;
    size_t base = ((size_t)b * cPIX + h * cKX) * cDIM + c;
    float acc = Hr[base];
    acc += ((w & 1) ? -1.0f : 1.0f) * Hr[base + (size_t)16 * cDIM];
    for (int k = 1; k < 16; ++k) {
        float ang = (PI2 / cW) * k * w;
        acc += 2.0f * (Hr[base + (size_t)k * cDIM] * __cosf(ang)
                     - Hi[base + (size_t)k * cDIM] * __sinf(ang));
    }
    sp[((size_t)b * cL + h * cW + w) * cDIM + c] = f2bf(acc * (1.0f / 32.0f));
}
// out = att + xc * sigmoid(glu), back to NHWC
__global__ void k_final(const float* __restrict__ att, const float* __restrict__ xnc,
                        const float* __restrict__ g, float* __restrict__ out) {
    int idx = blockIdx.x * blockDim.x + threadIdx.x;  // (b,l,c)
    if (idx >= cB * cL * cDIM) return;
    int c = idx & 127, l = (idx >> 7) & 1023, b = idx >> 17;
    size_t r = (size_t)b * cL + l;
    out[r * cDIM + c] = att[r * cDIM + c]
                      + xnc[((size_t)b * cDIM + c) * cL + l] * sigm(g[r * cDIM + c]);
}

// ---------------- host orchestration ----------------
static inline int cdiv(int a, int b) { return (a + b - 1) / b; }

extern "C" void kernel_launch(void* const* d_in, const int* in_sizes, int n_in,
                              void* d_out, int out_size, void* d_ws, size_t ws_size,
                              hipStream_t stream) {
    (void)in_sizes; (void)n_in; (void)out_size; (void)ws_size;
    const float* x       = (const float*)d_in[0];
    const float* mask    = (const float*)d_in[1];
    const float* ln1_w   = (const float*)d_in[2];
    const float* ln1_b   = (const float*)d_in[3];
    const float* ln2_w   = (const float*)d_in[4];
    const float* ln2_b   = (const float*)d_in[5];
    const float* in_w    = (const float*)d_in[6];
    const float* conv_w  = (const float*)d_in[7];
    const float* conv_b  = (const float*)d_in[8];
    const float* xproj_w = (const float*)d_in[9];
    const float* dt_w    = (const float*)d_in[10];
    const float* dt_b    = (const float*)d_in[11];
    const float* A_log   = (const float*)d_in[12];
    const float* Ds      = (const float*)d_in[13];
    const float* onw     = (const float*)d_in[14];
    const float* onb     = (const float*)d_in[15];
    const float* out_w   = (const float*)d_in[16];
    const float* glu_w   = (const float*)d_in[17];
    const float* glu_b   = (const float*)d_in[18];
    const float* mag_w1  = (const float*)d_in[19];
    const float* mag_b1  = (const float*)d_in[20];
    const float* mag_w2  = (const float*)d_in[21];
    const float* mag_b2  = (const float*)d_in[22];
    const float* pha_w1  = (const float*)d_in[23];
    const float* pha_b1  = (const float*)d_in[24];
    const float* pha_w2  = (const float*)d_in[25];
    const float* pha_b2  = (const float*)d_in[26];
    float* outp = (float*)d_out;

    size_t off = 0;
    auto alloc = [&](size_t bytes) -> char* {
        char* p = (char*)d_ws + off;
        off += (bytes + 255) & ~(size_t)255;
        return p;
    };
    const int M1 = cB * cL;                       // 4096 token rows

    // persistent region (weights + cross-phase activations)
    bf16_t* w_in  = (bf16_t*)alloc((size_t)2 * cDI * cDIM * 2);
    bf16_t* w_xp  = (bf16_t*)alloc((size_t)cK * cCP * cDI * 2);
    bf16_t* w_out = (bf16_t*)alloc((size_t)cDIM * cDI * 2);
    bf16_t* w_glu = (bf16_t*)alloc((size_t)cDIM * cDIM * 2);
    bf16_t* w_m1  = (bf16_t*)alloc((size_t)64 * cDIM * 2);
    bf16_t* w_m2  = (bf16_t*)alloc((size_t)cDIM * 64 * 2);
    bf16_t* w_p1  = (bf16_t*)alloc((size_t)64 * cDIM * 2);
    bf16_t* w_p2  = (bf16_t*)alloc((size_t)cDIM * 64 * 2);
    float*  att   = (float*)alloc((size_t)M1 * cDIM * 4);
    float*  xnc   = (float*)alloc((size_t)cB * cDIM * cL * 4);
    const size_t poolBase = off;

    // phase-1 pool (GASSM)
    bf16_t* xln  = (bf16_t*)alloc((size_t)M1 * cDIM * 2);
    float*  xz   = (float*)alloc((size_t)M1 * 2 * cDI * 4);
    float*  xc   = (float*)alloc((size_t)cB * cDI * cL * 4);
    bf16_t* xs   = (bf16_t*)alloc((size_t)cK * cB * cL * cDI * 2);
    float*  dbl  = (float*)alloc((size_t)cK * M1 * cCP * 4);
    float*  ys   = (float*)alloc((size_t)cK * cB * cL * cDI * 4);
    bf16_t* yact = (bf16_t*)alloc((size_t)M1 * cDI * 2);
    float*  ox   = (float*)alloc((size_t)M1 * cDIM * 4);

    // ---- weight conversion ----
    k_cvt_bf<<<cdiv(2 * cDI * cDIM, 256), 256, 0, stream>>>(in_w,  w_in,  2 * cDI * cDIM);
    k_cvt_xproj<<<cdiv(cK * cCP * cDI, 256), 256, 0, stream>>>(xproj_w, w_xp);
    k_cvt_bf<<<cdiv(cDIM * cDI, 256), 256, 0, stream>>>(out_w, w_out, cDIM * cDI);
    k_cvt_bf<<<cdiv(cDIM * cDIM, 256), 256, 0, stream>>>(glu_w, w_glu, cDIM * cDIM);
    k_cvt_bf<<<cdiv(64 * cDIM, 256), 256, 0, stream>>>(mag_w1, w_m1, 64 * cDIM);
    k_cvt_bf<<<cdiv(cDIM * 64, 256), 256, 0, stream>>>(mag_w2, w_m2, cDIM * 64);
    k_cvt_bf<<<cdiv(64 * cDIM, 256), 256, 0, stream>>>(pha_w1, w_p1, 64 * cDIM);
    k_cvt_bf<<<cdiv(cDIM * 64, 256), 256, 0, stream>>>(pha_w2, w_p2, cDIM * 64);

    // ---- GASSM ----
    k_ln_bf<cDIM><<<M1, cDIM, 0, stream>>>(x, ln1_w, ln1_b, xln);
    {   // in_proj: 4096 x 512 x 128 (16x64 strip per wave)
        int tiles = (M1 / 16) * (2 * cDI / 64);
        k_gemm_wmma<4><<<cdiv(tiles, 8), 256, 0, stream>>>(xln, w_in, nullptr, xz, M1, 2 * cDI, cDIM, 0);
    }
    k_dwconv<<<cdiv(cB * cDI * cL, 256), 256, 0, stream>>>(xz, conv_w, conv_b, mask, xc);
    k_build_xs<<<cdiv(cK * cB * cL * cDI, 256), 256, 0, stream>>>(xc, xs);
    for (int k = 0; k < cK; ++k) {   // x_proj per direction: 4096 x 48 x 256 (16x48 strip)
        int tiles = (M1 / 16) * (cCP / 48);
        k_gemm_wmma<3><<<cdiv(tiles, 8), 256, 0, stream>>>(
            xs + (size_t)k * cB * cL * cDI, w_xp + (size_t)k * cCP * cDI,
            nullptr, dbl + (size_t)k * M1 * cCP, M1, cCP, cDI, 0);
    }
    k_scan<<<cK * cB, cDI, 0, stream>>>(dbl, xs, dt_w, dt_b, A_log, Ds, ys);
    k_combine<<<M1, cDI, 0, stream>>>(ys, xz, onw, onb, yact);
    {   // out_proj: 4096 x 128 x 256
        int tiles = (M1 / 16) * (cDIM / 64);
        k_gemm_wmma<4><<<cdiv(tiles, 8), 256, 0, stream>>>(yact, w_out, nullptr, ox, M1, cDIM, cDI, 0);
    }
    k_add<<<cdiv(M1 * cDIM, 256), 256, 0, stream>>>(ox, x, att, M1 * cDIM);
    k_ln_t<<<M1, cDIM, 0, stream>>>(att, ln2_w, ln2_b, xnc);

    // ---- phase-2 pool (FDFG) reuses phase-1 pool (stream-ordered) ----
    off = poolBase;
    float*  Fr   = (float*)alloc((size_t)cB * cDIM * cH * cKX * 4);
    float*  Fi   = (float*)alloc((size_t)cB * cDIM * cH * cKX * 4);
    float*  mag  = (float*)alloc((size_t)cM2 * cDIM * 4);
    float*  pha  = (float*)alloc((size_t)cM2 * cDIM * 4);
    bf16_t* magb = (bf16_t*)alloc((size_t)cM2 * cDIM * 2);
    bf16_t* phab = (bf16_t*)alloc((size_t)cM2 * cDIM * 2);
    float*  tmag = (float*)alloc((size_t)cM2 * 64 * 4);
    float*  tpha = (float*)alloc((size_t)cM2 * 64 * 4);
    bf16_t* smag = (bf16_t*)alloc((size_t)cM2 * 64 * 2);
    bf16_t* spha = (bf16_t*)alloc((size_t)cM2 * 64 * 2);
    float*  magp = (float*)alloc((size_t)cM2 * cDIM * 4);
    float*  phap = (float*)alloc((size_t)cM2 * cDIM * 4);
    float*  F2r  = (float*)alloc((size_t)cM2 * cDIM * 4);
    float*  F2i  = (float*)alloc((size_t)cM2 * cDIM * 4);
    float*  Hr   = (float*)alloc((size_t)cM2 * cDIM * 4);
    float*  Hi   = (float*)alloc((size_t)cM2 * cDIM * 4);
    bf16_t* spb  = (bf16_t*)alloc((size_t)M1 * cDIM * 2);
    float*  g    = (float*)alloc((size_t)M1 * cDIM * 4);

    int nf = cB * cDIM * cH * cKX;   // 278528
    k_rfft_w<<<cdiv(nf, 256), 256, 0, stream>>>(xnc, Fr, Fi);
    k_fft_h_magpha<<<cdiv(nf, 256), 256, 0, stream>>>(Fr, Fi, mag, pha, magb, phab);
    {   // mag path: 2176x64x128 -> stdmean -> 2176x128x64
        int t1 = (cM2 / 16) * (64 / 64);
        k_gemm_wmma<4><<<cdiv(t1, 8), 256, 0, stream>>>(magb, w_m1, mag_b1, tmag, cM2, 64, cDIM, 2);
        k_stdmean<<<cdiv(cM2, 256), 256, 0, stream>>>(tmag, smag);
        int t2 = (cM2 / 16) * (cDIM / 64);
        k_gemm_wmma<4><<<cdiv(t2, 8), 256, 0, stream>>>(smag, w_m2, mag_b2, magp, cM2, cDIM, 64, 1);
    }
    {   // pha path
        int t1 = (cM2 / 16) * (64 / 64);
        k_gemm_wmma<4><<<cdiv(t1, 8), 256, 0, stream>>>(phab, w_p1, pha_b1, tpha, cM2, 64, cDIM, 2);
        k_stdmean<<<cdiv(cM2, 256), 256, 0, stream>>>(tpha, spha);
        int t2 = (cM2 / 16) * (cDIM / 64);
        k_gemm_wmma<4><<<cdiv(t2, 8), 256, 0, stream>>>(spha, w_p2, pha_b2, phap, cM2, cDIM, 64, 1);
    }
    k_recon<<<cdiv(cM2 * cDIM, 256), 256, 0, stream>>>(mag, magp, pha, phap, F2r, F2i);
    k_ifft_h<<<cdiv(nf, 256), 256, 0, stream>>>(F2r, F2i, Hr, Hi);
    k_irfft_w<<<cdiv(cB * cL * cDIM, 256), 256, 0, stream>>>(Hr, Hi, spb);
    {   // GLU conv: 4096 x 128 x 128
        int tiles = (M1 / 16) * (cDIM / 64);
        k_gemm_wmma<4><<<cdiv(tiles, 8), 256, 0, stream>>>(spb, w_glu, glu_b, g, M1, cDIM, cDIM, 1);
    }
    k_final<<<cdiv(cB * cL * cDIM, 256), 256, 0, stream>>>(att, xnc, g, outp);
}
